// HeposMultiHeadAttention_8392366096731
// MI455X (gfx1250) — matrix operand order repaired and checked by reading, not compiled, not run on hardware
//
#include <hip/hip_runtime.h>
#include <hip/hip_bf16.h>

typedef __attribute__((ext_vector_type(16))) _Float16 v16h;
typedef __attribute__((ext_vector_type(8)))  _Float16 v8h;
typedef __attribute__((ext_vector_type(4)))  _Float16 v4h;
typedef __attribute__((ext_vector_type(8)))  float    v8f;

union V16HU { v16h v; v8h h[2]; };

// ---------------------------------------------------------------------------
// WMMA fragment helpers (gfx1250, wave32).
// A fragment (16x32 f16, M x K): lane L holds row (L&15); halves 0..7 = K
// sel*8..sel*8+7, halves 8..15 = K 16+sel*8.., sel = L>>4.  (ISA 7.12.2)
// B fragment for C = A @ B^T with Bmat row-major [N][K]: column n = L&15,
// same K order as A: halves 0..7 = K sel*8.., halves 8..15 = K 16+sel*8..
// C/D 16x16 f32: element (M,N) in VGPR (M&7), lane N + 16*(M>>3).
// ---------------------------------------------------------------------------
static __device__ __forceinline__ v16h load_a_frag(const _Float16* A, size_t ld, int lane) {
  const int r = lane & 15, sel = lane >> 4;
  const _Float16* p = A + (size_t)r * ld + sel * 8;
  V16HU u;
  u.h[0] = *(const v8h*)(p);
  u.h[1] = *(const v8h*)(p + 16);
  return u.v;
}

static __device__ __forceinline__ v16h load_b_frag(const _Float16* Bm, size_t ld, int lane) {
  const int n = lane & 15, sel = lane >> 4;
  const _Float16* p = Bm + (size_t)n * ld + sel * 8;
  V16HU u;
  u.h[0] = *(const v8h*)(p);
  u.h[1] = *(const v8h*)(p + 16);
  return u.v;
}

static __device__ __forceinline__ v8f wmma_f16(v16h a, v16h b, v8f c) {
  return __builtin_amdgcn_wmma_f32_16x16x32_f16(false, a, false, b, (short)0, c, false, false);
}

// ---------------------------------------------------------------------------
// Max-reduce across each 16-lane half.  Prefer v_permlane16_b32 (pure VALU,
// no LDS pipeline / s_wait_dscnt round trip); fall back to ds_bpermute-based
// __shfl_xor if the builtin is unavailable.  Lane-select nibbles implement
// xor-1/2/4/8 within each group of 16 lanes (selector for lane i is nibble i
// of {src2,src1}).
// ---------------------------------------------------------------------------
#if __has_builtin(__builtin_amdgcn_permlane16)
#define ATTN_USE_PERMLANE 1
#endif

static __device__ __forceinline__ float grp16_max(float v) {
#ifdef ATTN_USE_PERMLANE
  unsigned x, y;
  x = __float_as_uint(v);
  y = __builtin_amdgcn_permlane16(x, x, 0x67452301u, 0xEFCDAB89u, false, false); // xor 1
  v = fmaxf(v, __uint_as_float(y));
  x = __float_as_uint(v);
  y = __builtin_amdgcn_permlane16(x, x, 0x54761032u, 0xDCFE98BAu, false, false); // xor 2
  v = fmaxf(v, __uint_as_float(y));
  x = __float_as_uint(v);
  y = __builtin_amdgcn_permlane16(x, x, 0x32107654u, 0xBA98FEDCu, false, false); // xor 4
  v = fmaxf(v, __uint_as_float(y));
  x = __float_as_uint(v);
  y = __builtin_amdgcn_permlane16(x, x, 0xFEDCBA98u, 0x76543210u, false, false); // xor 8
  v = fmaxf(v, __uint_as_float(y));
  return v;
#else
  v = fmaxf(v, __shfl_xor(v, 1, 32));
  v = fmaxf(v, __shfl_xor(v, 2, 32));
  v = fmaxf(v, __shfl_xor(v, 4, 32));
  v = fmaxf(v, __shfl_xor(v, 8, 32));
  return v;
#endif
}

// ---------------------------------------------------------------------------
// f32 -> f16 conversion (vectorized x4)
// ---------------------------------------------------------------------------
__global__ void cvt_f32_f16(const float* __restrict__ src, _Float16* __restrict__ dst, int n4) {
  int i = blockIdx.x * blockDim.x + threadIdx.x;
  if (i < n4) {
    float4 f = ((const float4*)src)[i];
    v4h h = {(_Float16)f.x, (_Float16)f.y, (_Float16)f.z, (_Float16)f.w};
    *(v4h*)(dst + (size_t)i * 4) = h;
  }
}

// ---------------------------------------------------------------------------
// Projection GEMM (1024x1024x1024): C = (A @ W^T + bias) * SCALE
// block = 256 threads = 8 waves, wave tile 32x64, WG tile 64x256.
// ---------------------------------------------------------------------------
template <bool F32OUT>
__global__ void __launch_bounds__(256) proj_gemm_kernel(
    const _Float16* __restrict__ A, const _Float16* __restrict__ W,
    const float* __restrict__ bias, _Float16* __restrict__ C16,
    float* __restrict__ C32, float scale)
{
  constexpr int N = 1024, K = 1024;
  const int lane = threadIdx.x & 31;
  const int w    = threadIdx.x >> 5;
  const int wm = w >> 2, wn = w & 3;
  const int row0 = blockIdx.y * 64 + wm * 32;
  const int col0 = blockIdx.x * 256 + wn * 64;

  v8f acc[2][4] = {};
  for (int kb = 0; kb < K; kb += 32) {
    v16h a[2], b[4];
#pragma unroll
    for (int mi = 0; mi < 2; ++mi)
      a[mi] = load_a_frag(A + (size_t)(row0 + mi * 16) * K + kb, K, lane);
#pragma unroll
    for (int ni = 0; ni < 4; ++ni)
      b[ni] = load_b_frag(W + (size_t)(col0 + ni * 16) * K + kb, K, lane);
#pragma unroll
    for (int mi = 0; mi < 2; ++mi)
#pragma unroll
      for (int ni = 0; ni < 4; ++ni)
        acc[mi][ni] = wmma_f16(a[mi], b[ni], acc[mi][ni]);
  }

  const int hf = lane >> 4, nl = lane & 15;
#pragma unroll
  for (int mi = 0; mi < 2; ++mi)
#pragma unroll
    for (int ni = 0; ni < 4; ++ni) {
      const int col = col0 + ni * 16 + nl;
      const float bv = bias[col];
      const int rbase = row0 + mi * 16 + 8 * hf;
#pragma unroll
      for (int r = 0; r < 8; ++r) {
        const float vv = (acc[mi][ni][r] + bv) * scale;
        if (F32OUT) C32[(size_t)(rbase + r) * N + col] = vv;
        else        C16[(size_t)(rbase + r) * N + col] = (_Float16)vv;
      }
    }
}

// ---------------------------------------------------------------------------
// HEPOS-compacted K/V projection.  Head h uses key rows {h, h+16, ...} (512)
// and weight rows [h*64, h*64+64).  TRANSPOSED=false: [16][512][64] f16 (K).
// TRANSPOSED=true (V): [16][64][512] f16 so P@V B-fragments are contiguous.
// grid = (16 heads, 2), block = 256 (8 waves x 32 keys).
// ---------------------------------------------------------------------------
template <bool TRANSPOSED>
__global__ void __launch_bounds__(256) kv_proj_kernel(
    const _Float16* __restrict__ X,   // [8192][1024]
    const _Float16* __restrict__ Wt,  // [1024][1024]
    const float* __restrict__ bias,   // [1024]
    _Float16* __restrict__ Out)
{
  constexpr int K = 1024;
  const int lane = threadIdx.x & 31;
  const int w    = threadIdx.x >> 5;
  const int h    = blockIdx.x;
  const int keybase = (blockIdx.y * 8 + w) * 32;

  v8f acc[2][4] = {};
  for (int kb = 0; kb < K; kb += 32) {
    v16h a[2], b[4];
#pragma unroll
    for (int mi = 0; mi < 2; ++mi)   // gathered rows: stride-16 keys -> ld = 16*1024
      a[mi] = load_a_frag(X + (size_t)(h + 16 * (keybase + mi * 16)) * 1024 + kb,
                          (size_t)16 * 1024, lane);
#pragma unroll
    for (int ni = 0; ni < 4; ++ni)
      b[ni] = load_b_frag(Wt + (size_t)(h * 64 + ni * 16) * K + kb, K, lane);
#pragma unroll
    for (int mi = 0; mi < 2; ++mi)
#pragma unroll
      for (int ni = 0; ni < 4; ++ni)
        acc[mi][ni] = wmma_f16(a[mi], b[ni], acc[mi][ni]);
  }

  const int hf = lane >> 4, nl = lane & 15;
#pragma unroll
  for (int mi = 0; mi < 2; ++mi)
#pragma unroll
    for (int ni = 0; ni < 4; ++ni) {
      const int col = ni * 16 + nl;                 // d within head
      const float bv = bias[h * 64 + col];
      const int rbase = keybase + mi * 16 + 8 * hf; // compact key idx
#pragma unroll
      for (int r = 0; r < 8; ++r) {
        const float vv = acc[mi][ni][r] + bv;
        if (TRANSPOSED)
          Out[(size_t)h * 64 * 512 + (size_t)col * 512 + (rbase + r)] = (_Float16)vv;
        else
          Out[(size_t)h * 512 * 64 + (size_t)(rbase + r) * 64 + col] = (_Float16)vv;
      }
    }
}

// ---------------------------------------------------------------------------
// Per-head flash attention over the 512 compacted keys (no mask needed).
// grid = (16 heads, 4), block = 256 = 8 waves; each wave owns 32 queries and
// loops over 8 key blocks of 64.  Q16 already carries the 1/sqrt(64) scale.
// Row max uses permlane16 (VALU) reductions within lane-halves.  Row sums are
// computed by WMMA itself: an extra accumulator tile P @ ones obeys the same
// flash recurrence  l_new = l*f + rowsum(P)  as  O_new = O*f + P@V, and since
// the ones-B fills all 16 columns, every lane holds its rows' sums directly.
// P is staged through LDS to convert C-layout -> A-fragment layout.
// ---------------------------------------------------------------------------
__global__ void __launch_bounds__(256) attn_kernel(
    const _Float16* __restrict__ Q16,  // [1024][1024], pre-scaled
    const _Float16* __restrict__ Kc,   // [16][512][64]
    const _Float16* __restrict__ VcT,  // [16][64][512]
    _Float16* __restrict__ O16)        // [1024][1024]
{
  __shared__ _Float16 Plds[8][32 * 64];   // 4KB per wave, 32KB total
  const int lane = threadIdx.x & 31;
  const int w    = threadIdx.x >> 5;
  const int h    = blockIdx.x;
  const int q0   = (blockIdx.y * 8 + w) * 32;
  const _Float16* Kh = Kc + (size_t)h * 512 * 64;
  const _Float16* Vh = VcT + (size_t)h * 64 * 512;
  _Float16* myP = Plds[w];
  const int hf = lane >> 4, nl = lane & 15;

  // constant all-ones B fragment for the P @ ones row-sum accumulator
  v16h onesf;
#pragma unroll
  for (int i = 0; i < 16; ++i) onesf[i] = (_Float16)1.0f;

  // Preload Q fragments [mi][ks] for the whole key loop (head dim 64 = 2 K-steps)
  v16h qa[2][2];
#pragma unroll
  for (int mi = 0; mi < 2; ++mi)
#pragma unroll
    for (int ks = 0; ks < 2; ++ks)
      qa[mi][ks] = load_a_frag(Q16 + (size_t)(q0 + mi * 16) * 1024 + h * 64 + ks * 32,
                               1024, lane);

  v8f o[2][4] = {};
  v8f osum[2] = {};          // running softmax denominators (all lanes)
  float m[2][8];
#pragma unroll
  for (int mi = 0; mi < 2; ++mi)
#pragma unroll
    for (int r = 0; r < 8; ++r) m[mi][r] = -__builtin_inff();

  for (int kb = 0; kb < 8; ++kb) {
    // ---- S = Qs @ Kc^T for this 32x64 block ------------------------------
    v8f s[2][4] = {};
#pragma unroll
    for (int ks = 0; ks < 2; ++ks) {
      v16h b[4];
#pragma unroll
      for (int ni = 0; ni < 4; ++ni)
        b[ni] = load_b_frag(Kh + (size_t)(kb * 64 + ni * 16) * 64 + ks * 32, 64, lane);
#pragma unroll
      for (int mi = 0; mi < 2; ++mi)
#pragma unroll
        for (int ni = 0; ni < 4; ++ni)
          s[mi][ni] = wmma_f16(qa[mi][ks], b[ni], s[mi][ni]);
    }

    // ---- online softmax update (max only; sums handled by WMMA) ----------
#pragma unroll
    for (int mi = 0; mi < 2; ++mi) {
#pragma unroll
      for (int r = 0; r < 8; ++r) {
        float rmax = fmaxf(fmaxf(s[mi][0][r], s[mi][1][r]),
                           fmaxf(s[mi][2][r], s[mi][3][r]));
        rmax = grp16_max(rmax);
        const float nm = fmaxf(m[mi][r], rmax);
        const float f  = __expf(m[mi][r] - nm);
        m[mi][r] = nm;
        osum[mi][r] *= f;
#pragma unroll
        for (int ni = 0; ni < 4; ++ni) {
          const float p = __expf(s[mi][ni][r] - nm);
          myP[(mi * 16 + 8 * hf + r) * 64 + ni * 16 + nl] = (_Float16)p;
          o[mi][ni][r] *= f;
        }
      }
    }
    asm volatile("s_wait_dscnt 0x0" ::: "memory");  // P stores visible to LDS reads

    // ---- O += P @ V, denominators += P @ ones ----------------------------
#pragma unroll
    for (int ks = 0; ks < 2; ++ks) {
      v16h a[2], b[4];
#pragma unroll
      for (int mi = 0; mi < 2; ++mi)
        a[mi] = load_a_frag(myP + (mi * 16) * 64 + ks * 32, 64, lane);
#pragma unroll
      for (int ni = 0; ni < 4; ++ni)
        b[ni] = load_b_frag(Vh + (size_t)(ni * 16) * 512 + kb * 64 + ks * 32, 512, lane);
#pragma unroll
      for (int mi = 0; mi < 2; ++mi) {
#pragma unroll
        for (int ni = 0; ni < 4; ++ni)
          o[mi][ni] = wmma_f16(a[mi], b[ni], o[mi][ni]);
        osum[mi] = wmma_f16(a[mi], onesf, osum[mi]);
      }
    }
  }

  // ---- finalize: divide by row sums, write head-interleaved O ------------
#pragma unroll
  for (int mi = 0; mi < 2; ++mi)
#pragma unroll
    for (int r = 0; r < 8; ++r) {
      const float inv = 1.0f / osum[mi][r];
      const int row = q0 + mi * 16 + 8 * hf + r;
#pragma unroll
      for (int ni = 0; ni < 4; ++ni) {
        const int col = h * 64 + ni * 16 + nl;
        O16[(size_t)row * 1024 + col] = (_Float16)(o[mi][ni][r] * inv);
      }
    }
}

// ---------------------------------------------------------------------------
extern "C" void kernel_launch(void* const* d_in, const int* in_sizes, int n_in,
                              void* d_out, int out_size, void* d_ws, size_t ws_size,
                              hipStream_t stream) {
  const float* q  = (const float*)d_in[0];
  const float* k  = (const float*)d_in[1];
  const float* v  = (const float*)d_in[2];
  const float* Wq = (const float*)d_in[3];
  const float* bq = (const float*)d_in[4];
  const float* Wk = (const float*)d_in[5];
  const float* bk = (const float*)d_in[6];
  const float* Wv = (const float*)d_in[7];
  const float* bv = (const float*)d_in[8];
  const float* Wo = (const float*)d_in[9];
  const float* bo = (const float*)d_in[10];
  float* out = (float*)d_out;

  // workspace carve-up (f16 elements); total ~24.5M halves = ~49MB
  _Float16* ws = (_Float16*)d_ws;
  size_t off = 0;
  auto carve = [&](size_t n) { _Float16* p = ws + off; off += n; return p; };
  _Float16* q16  = carve((size_t)1024 * 1024);
  _Float16* k16  = carve((size_t)8192 * 1024);
  _Float16* v16  = carve((size_t)8192 * 1024);
  _Float16* Wq16 = carve((size_t)1024 * 1024);
  _Float16* Wk16 = carve((size_t)1024 * 1024);
  _Float16* Wv16 = carve((size_t)1024 * 1024);
  _Float16* Wo16 = carve((size_t)1024 * 1024);
  _Float16* Q16  = carve((size_t)1024 * 1024);
  _Float16* Kc   = carve((size_t)16 * 512 * 64);
  _Float16* VcT  = carve((size_t)16 * 512 * 64);
  _Float16* O16  = carve((size_t)1024 * 1024);

  auto cvt = [&](const float* s, _Float16* d, size_t n) {
    int n4 = (int)(n / 4);
    cvt_f32_f16<<<(n4 + 255) / 256, 256, 0, stream>>>(s, d, n4);
  };
  cvt(q,  q16,  (size_t)1024 * 1024);
  cvt(k,  k16,  (size_t)8192 * 1024);
  cvt(v,  v16,  (size_t)8192 * 1024);
  cvt(Wq, Wq16, (size_t)1024 * 1024);
  cvt(Wk, Wk16, (size_t)1024 * 1024);
  cvt(Wv, Wv16, (size_t)1024 * 1024);
  cvt(Wo, Wo16, (size_t)1024 * 1024);

  // Q projection with folded softmax scale: Q16 = (q @ Wq^T + bq) / sqrt(64)
  proj_gemm_kernel<false><<<dim3(4, 16), 256, 0, stream>>>(q16, Wq16, bq, Q16,
                                                           nullptr, 0.125f);
  // HEPOS-compacted K/V projections (per head: 512 keys x 64 dims)
  kv_proj_kernel<false><<<dim3(16, 2), 256, 0, stream>>>(k16, Wk16, bk, Kc);
  kv_proj_kernel<true ><<<dim3(16, 2), 256, 0, stream>>>(v16, Wv16, bv, VcT);
  // attention (per head, flash softmax over 512 compacted keys)
  attn_kernel<<<dim3(16, 4), 256, 0, stream>>>(Q16, Kc, VcT, O16);
  // output projection: out = O @ Wo^T + bo   (f32 out)
  proj_gemm_kernel<true><<<dim3(4, 16), 256, 0, stream>>>(O16, Wo16, bo, nullptr,
                                                          out, 1.0f);
}
